// MultiHeadCrossAttention_7791070675855
// MI455X (gfx1250) — matrix operand order repaired and checked by reading, not compiled, not run on hardware
//
#include <hip/hip_runtime.h>
#include <hip/hip_bf16.h>

// ---------------------------------------------------------------------------
// MultiHeadCrossAttention for MI455X (gfx1250, wave32, WMMA bf16 16x16x32)
// B=16, N=1024, L=120, C=1024, H=16, Dh=64
// ---------------------------------------------------------------------------

#define BDIM   16
#define NDIM   1024
#define LDIM   120
#define LPAD   128
#define CDIM   1024
#define HEADS  16
#define DH     64

// LDS W-tile row pitch in bf16 elements (80 bytes): keeps 16B alignment for
// b128 reads and spreads the 16 column-rows over 16 distinct bank groups.
#define WPITCH 40

typedef __bf16 bf16;
typedef __attribute__((ext_vector_type(16))) __bf16 v16bf;
typedef __attribute__((ext_vector_type(8)))  __bf16 v8bf;
typedef __attribute__((ext_vector_type(8)))  float  v8f;

// round-to-nearest-even f32 -> bf16
__device__ __forceinline__ bf16 f2bf(float f) {
    union { unsigned short u; bf16 b; } cv;
    unsigned int x = __float_as_uint(f);
    cv.u = (unsigned short)((x + 0x7FFFu + ((x >> 16) & 1u)) >> 16);
    return cv.b;
}

__device__ __forceinline__ v16bf cat8(v8bf lo, v8bf hi) {
    return __builtin_shufflevector(lo, hi, 0,1,2,3,4,5,6,7,8,9,10,11,12,13,14,15);
}

// A-fragment (16x32 bf16): lane ln=lane&15 is row; g=lane>>4 selects k-half.
// element e: k = g*16 + e relative to k0
// => two contiguous 16B loads from the lane's row.
__device__ __forceinline__ v16bf load_a_frag(const bf16* __restrict__ row, int k0, int g) {
    const v8bf* p = reinterpret_cast<const v8bf*>(row);
    v8bf lo = p[(k0 + g * 16) >> 3];
    v8bf hi = p[(k0 + g * 16 + 8) >> 3];
    return cat8(lo, hi);
}

// B-fragment for B[k][n] where column n is contiguous in k (row of W / row of Vt):
// element e: value = wrow[k0 + g*16 + e] -> 16 contiguous halves (two 16B loads).
__device__ __forceinline__ v16bf load_bT_frag(const bf16* __restrict__ wrow, int k0, int g) {
    const v8bf* p = reinterpret_cast<const v8bf*>(wrow);
    int idx = (k0 + g * 16) >> 3;
    return cat8(p[idx], p[idx + 1]);
}

__device__ __forceinline__ v8f wmma_bf16(v16bf a, v16bf b, v8f c) {
    return __builtin_amdgcn_wmma_f32_16x16x32_bf16(
        /*neg_a=*/false, a, /*neg_b=*/false, b,
        /*c_mod=*/(short)0, c, /*reuse_a=*/false, /*reuse_b=*/false);
}

// CDNA5 async copy: per-lane 16B global -> LDS, tracked by ASYNCcnt.
__device__ __forceinline__ void async_copy_b128(unsigned lds_off, const void* g) {
    asm volatile("global_load_async_to_lds_b128 %0, %1, off"
                 :: "v"(lds_off), "v"(g)
                 : "memory");
}
__device__ __forceinline__ void wait_async_le2() {
    asm volatile("s_wait_asynccnt 0x2" ::: "memory");
}
__device__ __forceinline__ void wait_async_0() {
    asm volatile("s_wait_asynccnt 0x0" ::: "memory");
}

// ---------------------------------------------------------------------------
// fp32 -> bf16 conversion, 8 elements / thread / iteration
// ---------------------------------------------------------------------------
__global__ void f32_to_bf16_kernel(const float* __restrict__ in,
                                   bf16* __restrict__ out, long long n8) {
    long long i = (long long)blockIdx.x * blockDim.x + threadIdx.x;
    long long stride = (long long)gridDim.x * blockDim.x;
    for (; i < n8; i += stride) {
        const float4* p = reinterpret_cast<const float4*>(in + i * 8);
        float4 a = p[0], b = p[1];
        v8bf v;
        v[0] = f2bf(a.x); v[1] = f2bf(a.y); v[2] = f2bf(a.z); v[3] = f2bf(a.w);
        v[4] = f2bf(b.x); v[5] = f2bf(b.y); v[6] = f2bf(b.z); v[7] = f2bf(b.w);
        *reinterpret_cast<v8bf*>(out + i * 8) = v;
    }
}

// ---------------------------------------------------------------------------
// GEMM: out[m,n] = sum_k A[m,k] * W[n,k] + bias[n]
// A: MxK bf16 row-major, W: NcxK bf16 row-major.
// Block = 256 threads = 8 waves arranged 4(row)x2(col); block tile 128x128,
// wave tile 32x64.  W k-slices (128x32) are double-buffered in LDS via
// GLOBAL_LOAD_ASYNC_TO_LDS_B128 (ASYNCcnt), shared by all 8 waves.
// grid = (Nc/128, M/128)  -- N fastest so A rows stay hot in L2.
// ---------------------------------------------------------------------------
__device__ __forceinline__ void issue_wtile(const bf16* __restrict__ W, int K,
                                            int nbase, int k0,
                                            unsigned ldsBase, int tid) {
#pragma unroll
    for (int j = 0; j < 2; ++j) {
        int c   = tid + j * 256;        // 512 x 16B chunks per 128x32 tile
        int row = c >> 2;               // W row (n) within tile
        int seg = c & 3;                // 16B segment within the 64B k-slice
        unsigned loff = ldsBase + (unsigned)(row * (WPITCH * 2) + seg * 16);
        const bf16* g = W + (size_t)(nbase + row) * K + (k0 + seg * 8);
        async_copy_b128(loff, g);
    }
}

template <int OUT_BF16>
__global__ void __launch_bounds__(256)
gemm_bias(const bf16* __restrict__ A, const bf16* __restrict__ W,
          const float* __restrict__ bias, void* __restrict__ out,
          int M, int Nc, int K) {
    __shared__ __align__(16) bf16 Wt[2][128 * WPITCH];

    const int tid  = threadIdx.x;
    const int lane = tid & 31;
    const int wave = tid >> 5;
    const int g  = lane >> 4;
    const int ln = lane & 15;

    const int n_block = blockIdx.x * 128;
    const int m_block = blockIdx.y * 128;
    const int m0   = m_block + (wave >> 1) * 32;   // wave's 32 rows
    const int nloc = (wave & 1) * 64;              // wave's 64 cols (in tile)

    const unsigned ldsBase0 = (unsigned)(size_t)(&Wt[0][0]);
    const unsigned ldsBase1 = (unsigned)(size_t)(&Wt[1][0]);

    const int nk = K >> 5;
    issue_wtile(W, K, n_block, 0,  ldsBase0, tid);
    issue_wtile(W, K, n_block, 32, ldsBase1, tid);

    const bf16* arow0 = A + (size_t)(m0 + ln) * K;
    const bf16* arow1 = A + (size_t)(m0 + 16 + ln) * K;

    v8f acc[2][4] = {};

    for (int ks = 0; ks < nk; ++ks) {
        if (ks + 1 < nk) wait_async_le2();   // tile ks landed (tile ks+1 in flight)
        else             wait_async_0();
        __syncthreads();                      // all waves' copies visible

        const int k0 = ks << 5;
        const bf16* tile = &Wt[ks & 1][0];
        v16bf a0 = load_a_frag(arow0, k0, g);
        v16bf a1 = load_a_frag(arow1, k0, g);
#pragma unroll
        for (int t = 0; t < 4; ++t) {
            const v8bf* p = reinterpret_cast<const v8bf*>(
                tile + (nloc + t * 16 + ln) * WPITCH + g * 16);
            v16bf bfrag = cat8(p[0], p[1]);
            acc[0][t] = wmma_bf16(a0, bfrag, acc[0][t]);
            acc[1][t] = wmma_bf16(a1, bfrag, acc[1][t]);
        }
        __syncthreads();                      // everyone done reading buffer
        if (ks + 2 < nk)
            issue_wtile(W, K, n_block, (ks + 2) << 5,
                        (ks & 1) ? ldsBase1 : ldsBase0, tid);
    }

#pragma unroll
    for (int t = 0; t < 4; ++t) {
        const int col = n_block + nloc + t * 16 + ln;
        const float bv = bias[col];
#pragma unroll
        for (int rt = 0; rt < 2; ++rt) {
#pragma unroll
            for (int r = 0; r < 8; ++r) {
                const int row = m0 + rt * 16 + r + 8 * g;
                const float v = acc[rt][t][r] + bv;
                if (OUT_BF16)
                    ((bf16*)out)[(size_t)row * Nc + col] = f2bf(v);
                else
                    ((float*)out)[(size_t)row * Nc + col] = v;
            }
        }
    }
}

// ---------------------------------------------------------------------------
// Build Vt[b][h][d][l] (l padded to LPAD, pad = 0) from KV = [B*L, 2C] bf16,
// where V(b,l,h,d) = KV[b*L+l][C + h*DH + d].
// ---------------------------------------------------------------------------
__global__ void build_vt_kernel(const bf16* __restrict__ KVh, bf16* __restrict__ Vt) {
    int idx = blockIdx.x * blockDim.x + threadIdx.x;
    const int total = BDIM * HEADS * DH * LPAD;
    if (idx >= total) return;
    int l = idx % LPAD;
    int d = (idx / LPAD) % DH;
    int h = (idx / (LPAD * DH)) % HEADS;
    int b = idx / (LPAD * DH * HEADS);
    bf16 v = f2bf(0.0f);
    if (l < LDIM)
        v = KVh[(size_t)(b * LDIM + l) * (2 * CDIM) + CDIM + h * DH + d];
    Vt[idx] = v;
}

// ---------------------------------------------------------------------------
// Attention: per (b,h), per 16-row q tile (one wave each, 4 waves/block):
//   S = scale * Q Kt (masked) -> LDS f32 16x128
//   softmax over L (2 lanes/row + shfl_xor 16)
//   P bf16 16x128 in LDS; O = P V ; store bf16 * (1/rowsum)
// grid = B*H*(N/64), block = 128
// ---------------------------------------------------------------------------
__global__ void __launch_bounds__(128)
attention_kernel(const bf16* __restrict__ Qh, const bf16* __restrict__ KVh,
                 const bf16* __restrict__ Vt, const int* __restrict__ mask,
                 bf16* __restrict__ AOh) {
    const int lane = threadIdx.x & 31;
    const int wave = threadIdx.x >> 5;
    const int g  = lane >> 4;
    const int ln = lane & 15;

    const int nTiles = NDIM / 64;
    int t = blockIdx.x;
    const int nblk = t % nTiles;
    const int h = (t / nTiles) % HEADS;
    const int b = t / (nTiles * HEADS);
    const int m0 = nblk * 64 + wave * 16;   // query row base within N

    __shared__ __align__(16) float S[4][16][LPAD];
    __shared__ __align__(16) bf16  P[4][16][LPAD];
    __shared__ float Rinv[4][16];

    const float scale = 0.125f;             // Dh^-0.5, Dh=64

    // Q fragments (Dh = 64 -> two 16x32 k-steps), held for all 8 L-tiles
    const bf16* qrow = Qh + (size_t)(b * NDIM + m0 + ln) * CDIM + h * DH;
    const v16bf qa0 = load_a_frag(qrow, 0, g);
    const v16bf qa1 = load_a_frag(qrow, 32, g);

    // ---- scores ----
#pragma unroll
    for (int lt = 0; lt < LPAD / 16; ++lt) {
        const int lcol = lt * 16 + ln;
        const int lclamp = lcol < LDIM ? lcol : (LDIM - 1);
        const bf16* krow = KVh + (size_t)(b * LDIM + lclamp) * (2 * CDIM) + h * DH;
        v8f s = {};
        s = wmma_bf16(qa0, load_bT_frag(krow, 0, g), s);
        s = wmma_bf16(qa1, load_bT_frag(krow, 32, g), s);
        const bool valid = (lcol < LDIM) && (mask[b * LDIM + lcol] != 0);
#pragma unroll
        for (int r = 0; r < 8; ++r) {
            float v = valid ? s[r] * scale : -__builtin_inff();
            S[wave][r + 8 * g][lt * 16 + ln] = v;
        }
    }
    __syncthreads();

    // ---- softmax (lane handles row=ln, cols g*64..g*64+63) ----
    {
        const int row = ln;
        float mx = -__builtin_inff();
#pragma unroll 8
        for (int c = 0; c < 64; ++c) mx = fmaxf(mx, S[wave][row][g * 64 + c]);
        mx = fmaxf(mx, __shfl_xor(mx, 16, 32));
        float sum = 0.0f;
#pragma unroll 8
        for (int c = 0; c < 64; ++c) {
            float e = __expf(S[wave][row][g * 64 + c] - mx);
            sum += e;
            P[wave][row][g * 64 + c] = f2bf(e);
        }
        sum += __shfl_xor(sum, 16, 32);
        if (g == 0) Rinv[wave][row] = 1.0f / sum;
    }
    __syncthreads();

    // ---- O = P @ V ----
    const bf16* vbase = Vt + (size_t)(b * HEADS + h) * DH * LPAD;
    v8f o[4] = {};
#pragma unroll
    for (int k0 = 0; k0 < LPAD; k0 += 32) {
        v16bf pa = load_a_frag((const bf16*)&P[wave][ln][0], k0, g);
#pragma unroll
        for (int dt = 0; dt < 4; ++dt) {
            const bf16* vrow = vbase + (size_t)(dt * 16 + ln) * LPAD;
            o[dt] = wmma_bf16(pa, load_bT_frag(vrow, k0, g), o[dt]);
        }
    }
#pragma unroll
    for (int dt = 0; dt < 4; ++dt) {
#pragma unroll
        for (int r = 0; r < 8; ++r) {
            const int m = r + 8 * g;
            const float v = o[dt][r] * Rinv[wave][m];
            AOh[(size_t)(b * NDIM + m0 + m) * CDIM + h * DH + dt * 16 + ln] = f2bf(v);
        }
    }
}

// ---------------------------------------------------------------------------
// Launch
// ---------------------------------------------------------------------------
extern "C" void kernel_launch(void* const* d_in, const int* in_sizes, int n_in,
                              void* d_out, int out_size, void* d_ws, size_t ws_size,
                              hipStream_t stream) {
    (void)in_sizes; (void)n_in; (void)out_size; (void)ws_size;
    const float* x    = (const float*)d_in[0];
    const float* cond = (const float*)d_in[1];
    const int*   mask = (const int*)d_in[2];
    const float* Wq   = (const float*)d_in[3];
    const float* bq   = (const float*)d_in[4];
    const float* Wkv  = (const float*)d_in[5];
    const float* bkv  = (const float*)d_in[6];
    const float* Wp   = (const float*)d_in[7];
    const float* bp   = (const float*)d_in[8];

    char* ws = (char*)d_ws;
    size_t off = 0;
    auto alloc = [&](size_t bytes) {
        void* p = ws + off;
        off += (bytes + 255) & ~(size_t)255;
        return p;
    };
    bf16* Xh   = (bf16*)alloc((size_t)BDIM * NDIM * CDIM * 2);
    bf16* Ch   = (bf16*)alloc((size_t)BDIM * LDIM * CDIM * 2);
    bf16* WqH  = (bf16*)alloc((size_t)CDIM * CDIM * 2);
    bf16* WkvH = (bf16*)alloc((size_t)2 * CDIM * CDIM * 2);
    bf16* WpH  = (bf16*)alloc((size_t)CDIM * CDIM * 2);
    bf16* Qh   = (bf16*)alloc((size_t)BDIM * NDIM * CDIM * 2);
    bf16* KVh  = (bf16*)alloc((size_t)BDIM * LDIM * 2 * CDIM * 2);
    bf16* Vt   = (bf16*)alloc((size_t)BDIM * HEADS * DH * LPAD * 2);
    bf16* AOh  = (bf16*)alloc((size_t)BDIM * NDIM * CDIM * 2);

    auto cvt = [&](const float* in, bf16* outp, long long n) {
        long long n8 = n / 8;
        int blocks = (int)((n8 + 255) / 256);
        if (blocks > 2048) blocks = 2048;
        f32_to_bf16_kernel<<<blocks, 256, 0, stream>>>(in, outp, n8);
    };
    cvt(x,    Xh,   (long long)BDIM * NDIM * CDIM);
    cvt(cond, Ch,   (long long)BDIM * LDIM * CDIM);
    cvt(Wq,   WqH,  (long long)CDIM * CDIM);
    cvt(Wkv,  WkvH, (long long)2 * CDIM * CDIM);
    cvt(Wp,   WpH,  (long long)CDIM * CDIM);

    // Q = Xh * Wq^T + bq   (bf16 out)    grid (N/128, M/128), N fastest
    {
        dim3 grid(CDIM / 128, (BDIM * NDIM) / 128);
        gemm_bias<1><<<grid, 256, 0, stream>>>(Xh, WqH, bq, Qh,
                                               BDIM * NDIM, CDIM, CDIM);
    }
    // KV = Ch * Wkv^T + bkv (bf16 out) ; B*L = 1920 = 15*128
    {
        dim3 grid((2 * CDIM) / 128, (BDIM * LDIM) / 128);
        gemm_bias<1><<<grid, 256, 0, stream>>>(Ch, WkvH, bkv, KVh,
                                               BDIM * LDIM, 2 * CDIM, CDIM);
    }
    // V transpose (with zero pad to LPAD)
    {
        const int total = BDIM * HEADS * DH * LPAD;
        build_vt_kernel<<<(total + 255) / 256, 256, 0, stream>>>(KVh, Vt);
    }
    // attention
    attention_kernel<<<BDIM * HEADS * (NDIM / 64), 128, 0, stream>>>(
        Qh, KVh, Vt, mask, AOh);
    // out = AOh * Wp^T + bp (f32 out)
    {
        dim3 grid(CDIM / 128, (BDIM * NDIM) / 128);
        gemm_bias<0><<<grid, 256, 0, stream>>>(AOh, WpH, bp, (float*)d_out,
                                               BDIM * NDIM, CDIM, CDIM);
    }
}
